// My_model_7962869367095
// MI455X (gfx1250) — matrix operand order repaired and checked by reading, not compiled, not run on hardware
//
#include <hip/hip_runtime.h>
#include <hip/hip_bf16.h>
#include <math.h>

// Problem constants (from reference): E=100, H=64, B=64, T=20, K=50
#define E_DIM 100
#define EP 128          // padded E for WMMA K-chunks (4 x 32)
#define H_DIM 64
#define B_DIM 64
#define T_DIM 20
#define K_DIM 50

typedef __attribute__((ext_vector_type(16))) _Float16 v16h;
typedef __attribute__((ext_vector_type(8)))  _Float16 v8h;
typedef __attribute__((ext_vector_type(4)))  _Float16 v4h;
typedef __attribute__((ext_vector_type(8)))  float    v8f;

// ---------------------------------------------------------------------------
// WMMA helpers (V_WMMA_F32_16X16X32_F16, wave32). All operands come from LDS
// rows padded to 128 f16 columns (cols >= 100 are zero), so fragment loads are
// branch-free 16B vector loads.
// ---------------------------------------------------------------------------
__device__ __forceinline__ v16h frag_pair(const _Float16* p0, const _Float16* p1) {
  v8h a = *(const v8h*)p0;
  v8h b = *(const v8h*)p1;
  return __builtin_shufflevector(a, b, 0, 1, 2, 3, 4, 5, 6, 7,
                                 8, 9, 10, 11, 12, 13, 14, 15);
}

// A fragment (16x32): lanes 0-15 row M=lane K={0..7,16..23};
// lanes 16-31 row M=lane-16 K={8..15,24..31}. base = LDS row of this lane's M.
__device__ __forceinline__ v16h fragA_lds(const _Float16* base, int kc, int lane) {
  const int hi = (lane & 16) ? 8 : 0;
  const _Float16* p = base + kc * 32 + hi;
  return frag_pair(p, p + 16);
}

// B fragment (32x16): lanes 0-15 col N=lane K={0..7,16..23}; lanes 16-31 K={8..15,24..31}.
// base = LDS row holding column N's K-elements contiguously.
__device__ __forceinline__ v16h fragB_lds(const _Float16* base, int kc, int lane) {
  const _Float16* p = base + kc * 32 + ((lane & 16) ? 8 : 0);
  return frag_pair(p, p + 16);
}

__device__ __forceinline__ v8f wmma_f16(v16h a, v16h b, v8f c) {
  return __builtin_amdgcn_wmma_f32_16x16x32_f16(
      /*neg_a=*/false, a, /*neg_b=*/false, b,
      /*c_mod=*/(short)0, c, /*reuse_a=*/false, /*reuse_b=*/false);
}

// Stage one f32 row chunk (float4) into LDS f16 (packed 4-half store, 8B aligned)
__device__ __forceinline__ void stage_chunk(const float* __restrict__ rp, int chunk,
                                            _Float16* __restrict__ dstRow) {
  float4 v = *(const float4*)(rp + chunk * 4);
  v4h h = {(_Float16)v.x, (_Float16)v.y, (_Float16)v.z, (_Float16)v.w};
  *(v4h*)(dstRow + chunk * 4) = h;
}

__device__ __forceinline__ float log_sigmoid(float x) {
  return fminf(x, 0.0f) - log1pf(expf(-fabsf(x)));
}

__device__ __forceinline__ float wave_dot(const float* __restrict__ x,
                                          const float* __restrict__ y, int lane) {
  float s = 0.0f;
  for (int e = lane; e < E_DIM; e += 32) s += x[e] * y[e];
#pragma unroll
  for (int m = 16; m >= 1; m >>= 1) s += __shfl_xor(s, m);
  return s;
}

__device__ __forceinline__ float wave_sum(const float* __restrict__ x, int lane) {
  float s = 0.0f;
  for (int e = lane; e < E_DIM; e += 32) s += x[e];
#pragma unroll
  for (int m = 16; m >= 1; m >>= 1) s += __shfl_xor(s, m);
  return s;
}

// ---------------------------------------------------------------------------
// Kernel 1: p_l = sum_i (-0.5/sigma[i]) * sum((v - mu[i])^2) over gathered rows
// ---------------------------------------------------------------------------
__global__ void prior_kernel(const float* __restrict__ alpha, const float* __restrict__ rho,
                             const float* __restrict__ lam,   const float* __restrict__ beta,
                             const float* __restrict__ mu_tab,
                             const float* __restrict__ theta, const float* __restrict__ gamma,
                             const float* __restrict__ sigma_list,
                             const float* __restrict__ mu_list,
                             const int* __restrict__ itemset, const int* __restrict__ userid,
                             float* __restrict__ out) {
  const int NITEM = 5 * B_DIM * K_DIM * E_DIM;  // 1,600,000
  const int NUSER = 2 * B_DIM * E_DIM;          // 12,800
  int idx = blockIdx.x * blockDim.x + threadIdx.x;
  float acc = 0.0f;
  if (idx < NITEM + NUSER) {
    float v, sig, mu;
    if (idx < NITEM) {
      const int per = B_DIM * K_DIM * E_DIM;
      int ti = idx / per;
      int rem = idx - ti * per;
      int row = rem / E_DIM;
      int e = rem - row * E_DIM;
      int gi = itemset[row];
      const float* tp;
      switch (ti) {
        case 0: tp = alpha; break;
        case 1: tp = rho; break;
        case 2: tp = lam; break;
        case 3: tp = beta; break;
        default: tp = mu_tab; break;
      }
      v = tp[(size_t)gi * E_DIM + e];
      sig = sigma_list[ti];
      mu = mu_list[ti];
    } else {
      int r2 = idx - NITEM;
      const int per = B_DIM * E_DIM;
      int ti = r2 / per;
      int rem = r2 - ti * per;
      int row = rem / E_DIM;
      int e = rem - row * E_DIM;
      int u = userid[row];
      const float* tp = (ti == 0) ? theta : gamma;
      v = tp[(size_t)u * E_DIM + e];
      sig = sigma_list[5 + ti];
      mu = mu_list[5 + ti];
    }
    float d = v - mu;
    acc = (-0.5f / sig) * d * d;
  }
#pragma unroll
  for (int m = 16; m >= 1; m >>= 1) acc += __shfl_xor(acc, m);
  if ((threadIdx.x & 31) == 0) atomicAdd(out, acc);
}

// ---------------------------------------------------------------------------
// Kernel 2: q_l via WMMA MLP. 1008 wave-groups of 16 rows; 4 waves / block.
// LDS: w1 as f16 [64][128] (block-shared) + per-wave X tile [16][128] f16.
// ---------------------------------------------------------------------------
__global__ void __launch_bounds__(128)
qf_kernel(const float* __restrict__ alpha, const float* __restrict__ rho,
          const float* __restrict__ lam,   const float* __restrict__ beta,
          const float* __restrict__ mu_tab,
          const float* __restrict__ theta, const float* __restrict__ gamma,
          const float* __restrict__ w1, const float* __restrict__ b1,
          const float* __restrict__ w2, const float* __restrict__ b2,
          const int* __restrict__ itemset, const int* __restrict__ userid,
          float* __restrict__ out) {
  __shared__ __align__(16) _Float16 sW1[H_DIM * EP];       // 16 KB
  __shared__ __align__(16) _Float16 sX[4][16 * EP];        // 4 x 4 KB

  const int tid = threadIdx.x;
  const int lane = tid & 31;
  const int wid = tid >> 5;
  const int gwave = blockIdx.x * 4 + wid;   // grid is exactly 252*4 = 1008
  const int rl = lane & 15;

  // ---- this lane's source row pointer (row rl of the 16-row group) ----
  const float* rowp;
  if (gwave < 1000) {
    int ti = gwave / 200;
    int rowIdx = (gwave - ti * 200) * 16 + rl;   // 0..3199
    int gi = itemset[rowIdx];
    const float* tp;
    switch (ti) {
      case 0: tp = alpha; break;
      case 1: tp = rho; break;
      case 2: tp = lam; break;
      case 3: tp = beta; break;
      default: tp = mu_tab; break;
    }
    rowp = tp + (size_t)gi * E_DIM;
  } else if (gwave < 1004) {
    int r = (gwave - 1000) * 16 + rl;
    rowp = theta + (size_t)userid[r] * E_DIM;
  } else {
    int r = (gwave - 1004) * 16 + rl;
    rowp = gamma + (size_t)userid[r] * E_DIM;
  }
  __builtin_prefetch(rowp, 0, 0);   // -> global_prefetch_b8

  // ---- stage w1 (block-wide): 64 rows x 25 float4 chunks ----
  for (int idx = tid; idx < H_DIM * 25; idx += 128) {
    int h = idx / 25, c = idx - h * 25;
    stage_chunk(w1 + (size_t)h * E_DIM, c, sW1 + h * EP);
  }
  for (int idx = tid; idx < H_DIM * 28; idx += 128) {  // zero pad cols 100..127
    int h = idx / 28, j = idx - h * 28;
    sW1[h * EP + 100 + j] = (_Float16)0.0f;
  }

  // ---- stage this wave's 16 rows (pointers distributed via 64-bit shuffle) ----
  unsigned long long up = (unsigned long long)rowp;
  _Float16* xbase = &sX[wid][0];
  for (int idx = lane; idx < 16 * 25; idx += 32) {
    int row = idx / 25, c = idx - row * 25;
    const float* rp = (const float*)__shfl(up, row);
    stage_chunk(rp, c, xbase + row * EP);
  }
  for (int idx = lane; idx < 16 * 28; idx += 32) {
    int row = idx / 28, j = idx - row * 28;
    xbase[row * EP + 100 + j] = (_Float16)0.0f;
  }
  __syncthreads();

  // ---- WMMA MLP ----
  const _Float16* arow = xbase + rl * EP;   // this lane's M row
  float acc2[8] = {0.f, 0.f, 0.f, 0.f, 0.f, 0.f, 0.f, 0.f};
#pragma unroll
  for (int nt = 0; nt < 4; ++nt) {            // 4 x 16 hidden units
    int h = nt * 16 + rl;
    float bh = b1[h];
    v8f c;
#pragma unroll
    for (int r = 0; r < 8; ++r) c[r] = bh;    // bias as C init
    const _Float16* bcol = sW1 + h * EP;
#pragma unroll
    for (int kc = 0; kc < 4; ++kc) {
      v16h a = fragA_lds(arow, kc, lane);
      v16h bw = fragB_lds(bcol, kc, lane);
      c = wmma_f16(a, bw, c);
    }
    float w2h = w2[h];
#pragma unroll
    for (int r = 0; r < 8; ++r) acc2[r] += fmaxf(c[r], 0.0f) * w2h;
  }
#pragma unroll
  for (int m = 1; m < 16; m <<= 1) {
#pragma unroll
    for (int r = 0; r < 8; ++r) acc2[r] += __shfl_xor(acc2[r], m);
  }
  float s = 0.0f;
  if (rl == 0) {                 // lane 0: rows 0..7, lane 16: rows 8..15
    float bb = b2[0];
#pragma unroll
    for (int r = 0; r < 8; ++r) s += 1.0f / (1.0f + expf(-(acc2[r] + bb)));
  }
  float s_hi = __shfl(s, 16);
  if (lane == 0) atomicAdd(out, -(s + s_hi));
}

// ---------------------------------------------------------------------------
// Kernel 3: basket term. One block per batch element b (256 threads = 8 waves).
// f16 LDS row layout (stride EP=128, all padded with zeros beyond col 99):
//   rows [0,50)    a_c  = alpha[itemset]
//   rows [50,100)  r_c  = rho[itemset]
//   rows [100,120) a0   = alpha[item_ids]
//   rows [120,140) r0   = rho[item_ids]
//   row  140       all-zero row (stand-in for out-of-range tile rows/cols)
// ---------------------------------------------------------------------------
#define ZROW 140

__global__ void __launch_bounds__(256)
basket_kernel(const float* __restrict__ alpha, const float* __restrict__ rho,
              const float* __restrict__ lam,   const float* __restrict__ beta,
              const float* __restrict__ theta, const float* __restrict__ gamma,
              const float* __restrict__ prices, const float* __restrict__ price_table,
              const int* __restrict__ item_ids, const int* __restrict__ itemset,
              const int* __restrict__ userid,
              float* __restrict__ out) {
  __shared__ __align__(16) _Float16 sF16[141 * EP];   // ~36 KB
  __shared__ float sA[K_DIM * 64];                    // A[k][j]
  __shared__ float sRDA[T_DIM * 64];                  // rda0[t][j]
  __shared__ float sG[T_DIM * 64];                    // G[t][k]
  __shared__ float sKesc[K_DIM];
  __shared__ float sKes0[T_DIM];
  __shared__ float sG0[T_DIM];
  __shared__ int sItems[K_DIM];
  __shared__ int sIids[T_DIM];

  const int b = blockIdx.x;
  const int tid = threadIdx.x;
  const int lane = tid & 31;
  const int wid = tid >> 5;
  const int nWaves = blockDim.x >> 5;

  if (tid < K_DIM) sItems[tid] = itemset[b * K_DIM + tid];
  if (tid >= 64 && tid < 64 + T_DIM) sIids[tid - 64] = item_ids[b * T_DIM + (tid - 64)];
  __syncthreads();

  // ---- stage 140 gathered rows as f16 (vectorized) + zero padding ----
  for (int idx = tid; idx < 140 * 25; idx += 256) {
    int row = idx / 25, c = idx - row * 25;
    const float* rp;
    if (row < 50)       rp = alpha + (size_t)sItems[row] * E_DIM;
    else if (row < 100) rp = rho   + (size_t)sItems[row - 50] * E_DIM;
    else if (row < 120) rp = alpha + (size_t)sIids[row - 100] * E_DIM;
    else                rp = rho   + (size_t)sIids[row - 120] * E_DIM;
    stage_chunk(rp, c, sF16 + row * EP);
  }
  for (int idx = tid; idx < 141 * 28; idx += 256) {   // cols 100..127 of every row
    int row = idx / 28, j = idx - row * 28;
    sF16[row * EP + 100 + j] = (_Float16)0.0f;
  }
  for (int idx = tid; idx < 100; idx += 256)          // rest of the zero row
    sF16[ZROW * EP + idx] = (_Float16)0.0f;

  const int uid = userid[b];
  const float* thp = theta + (size_t)uid * E_DIM;
  const float* gap = gamma + (size_t)uid * E_DIM;

  // ---- kesc[j], kes0[t], g0[t] (global wave-dots; independent of staging) ----
  for (int task = wid; task < K_DIM + T_DIM; task += nWaves) {
    if (task < K_DIM) {
      int gi = sItems[task];
      float thA = wave_dot(thp, alpha + (size_t)gi * E_DIM, lane);
      float gaB = wave_dot(gap, beta + (size_t)gi * E_DIM, lane);
      float sl = wave_sum(lam + (size_t)gi * E_DIM, lane);
      if (lane == 0)
        sKesc[task] = (sl + thA - gaB * logf(price_table[gi])) / (float)E_DIM;
    } else {
      int t = task - K_DIM;
      int ii = sIids[t];
      float thA = wave_dot(thp, alpha + (size_t)ii * E_DIM, lane);
      float gaB = wave_dot(gap, beta + (size_t)ii * E_DIM, lane);
      float sl = wave_sum(lam + (size_t)ii * E_DIM, lane);
      float g0 = wave_dot(rho + (size_t)ii * E_DIM, alpha + (size_t)ii * E_DIM, lane);
      if (lane == 0) {
        sKes0[t] = (sl + thA - gaB * logf(prices[b * T_DIM + t])) / (float)E_DIM;
        sG0[t] = g0;
      }
    }
  }
  __syncthreads();

  // ---- 32 WMMA output tiles: 16 for A, 8 for rda0, 8 for G ----
  for (int tile = wid; tile < 32; tile += nWaves) {
    int mat, mt, nt;
    if (tile < 16)      { mat = 0; mt = tile >> 2;        nt = tile & 3; }
    else if (tile < 24) { mat = 1; mt = (tile - 16) >> 2; nt = tile & 3; }
    else                { mat = 2; mt = (tile - 24) >> 2; nt = tile & 3; }

    const int mloc = mt * 16 + (lane & 15);
    int arow;
    if (mat == 0)      arow = (mloc < K_DIM) ? mloc        : ZROW;   // a_c
    else if (mat == 1) arow = (mloc < T_DIM) ? 100 + mloc  : ZROW;   // a0
    else               arow = (mloc < T_DIM) ? 120 + mloc  : ZROW;   // r0

    const int ncol = nt * 16 + (lane & 15);
    int brow;
    if (mat == 2) brow = (ncol < K_DIM) ? ncol      : ZROW;          // a_c
    else          brow = (ncol < K_DIM) ? 50 + ncol : ZROW;          // r_c

    const _Float16* abase = sF16 + arow * EP;
    const _Float16* bbase = sF16 + brow * EP;

    v8f d = {0.f, 0.f, 0.f, 0.f, 0.f, 0.f, 0.f, 0.f};
#pragma unroll
    for (int kc = 0; kc < 4; ++kc) {
      v16h a = fragA_lds(abase, kc, lane);
      v16h bb = fragB_lds(bbase, kc, lane);
      d = wmma_f16(a, bb, d);
    }
    float* dst = (mat == 0) ? sA : ((mat == 1) ? sRDA : sG);
    const int vm = (mat == 0) ? K_DIM : T_DIM;
#pragma unroll
    for (int r = 0; r < 8; ++r) {
      int row = mt * 16 + ((lane >> 4) << 3) + r;
      int col = nt * 16 + (lane & 15);
      if (row < vm && col < K_DIM) dst[row * 64 + col] = d[r];
    }
  }
  __syncthreads();

  // ---- per-t sequential k loop; lanes hold j = lane and j + 32 ----
  float basket = 0.0f;
  for (int t = wid; t < T_DIM; t += nWaves) {
    const int iid = sIids[t];
    const int j0 = lane, j1 = lane + 32;
    const bool inr1 = (j1 < K_DIM);
    const bool mj0 = (sItems[j0] != iid);
    const bool mj1 = inr1 ? (sItems[j1] != iid) : false;
    const float rda_j0 = sRDA[t * 64 + j0];
    const float rda_j1 = inr1 ? sRDA[t * 64 + j1] : 0.0f;
    const float kc_j0 = sKesc[j0];
    const float kc_j1 = inr1 ? sKesc[j1] : 0.0f;
    const float g0t = sG0[t];
    const float kes0t = sKes0[t];

    float cs0 = 0.0f, cs1 = 0.0f, csg = 0.0f;
    float len = 1.0f;
    const float NEG = -1e30f;

    for (int k = 0; k < K_DIM; ++k) {
      const bool mk = (sItems[k] != iid);
      const float Ak0 = sA[k * 64 + j0];
      const float Ak1 = inr1 ? sA[k * 64 + j1] : 0.0f;
      if (mk) {
        cs0 += Ak0;
        cs1 += Ak1;
        csg += sG[t * 64 + k];
        len += 1.0f;
      }
      const float denom = (len + 1.0f) * (float)E_DIM;
      const float D0 = rda_j0 + cs0;
      const float D1 = rda_j1 + cs1;
      const float v00 = (D0 + rda_j0) / denom + kc_j0;
      const float v01 = (D1 + rda_j1) / denom + kc_j1;
      const float vc0 = (D0 + Ak0) / denom + kc_j0;
      const float vc1 = (D1 + Ak1) / denom + kc_j1;
      const bool e0 = mj0 && (j0 > k);
      const bool e1 = mj1 && (j1 > k);
      float m0 = fmaxf(e0 ? v00 : NEG, e1 ? v01 : NEG);
      float mc = fmaxf(e0 ? vc0 : NEG, e1 ? vc1 : NEG);
#pragma unroll
      for (int s = 16; s >= 1; s >>= 1) {
        m0 = fmaxf(m0, __shfl_xor(m0, s));
        mc = fmaxf(mc, __shfl_xor(mc, s));
      }
      const float look0 = fmaxf(0.0f, m0);
      const float lookc = fmaxf(0.0f, mc);
      const float interact0 = (g0t + csg) / ((float)E_DIM * len);
      const float csk = __shfl((k < 32) ? cs0 : cs1, k & 31);
      const float interactc = (sRDA[t * 64 + k] + csk) / ((float)E_DIM * len);
      const float Kes0 = kes0t + interact0 + look0;
      const float KesC = sKesc[k] + interactc + lookc;
      if (mk && lane == 0) basket += log_sigmoid(Kes0 - KesC);
    }
  }
  if (lane == 0) atomicAdd(out, basket);
}

// ---------------------------------------------------------------------------
// Host launcher
// ---------------------------------------------------------------------------
extern "C" void kernel_launch(void* const* d_in, const int* in_sizes, int n_in,
                              void* d_out, int out_size, void* d_ws, size_t ws_size,
                              hipStream_t stream) {
  const float* alpha = (const float*)d_in[0];
  const float* rho = (const float*)d_in[1];
  const float* lam = (const float*)d_in[2];
  const float* beta = (const float*)d_in[3];
  const float* mu_tab = (const float*)d_in[4];
  const float* theta = (const float*)d_in[5];
  const float* gamma = (const float*)d_in[6];
  const float* w1 = (const float*)d_in[7];
  const float* b1 = (const float*)d_in[8];
  const float* w2 = (const float*)d_in[9];
  const float* b2 = (const float*)d_in[10];
  const float* sigma_list = (const float*)d_in[11];
  const float* mu_list = (const float*)d_in[12];
  const float* prices = (const float*)d_in[13];
  const float* price_table = (const float*)d_in[14];
  const int* item_ids = (const int*)d_in[15];
  const int* itemset = (const int*)d_in[16];
  const int* userid = (const int*)d_in[17];
  float* out = (float*)d_out;

  hipMemsetAsync(out, 0, sizeof(float), stream);

  const int prior_total = 5 * B_DIM * K_DIM * E_DIM + 2 * B_DIM * E_DIM;
  prior_kernel<<<(prior_total + 255) / 256, 256, 0, stream>>>(
      alpha, rho, lam, beta, mu_tab, theta, gamma, sigma_list, mu_list,
      itemset, userid, out);

  qf_kernel<<<252, 128, 0, stream>>>(alpha, rho, lam, beta, mu_tab, theta, gamma,
                                     w1, b1, w2, b2, itemset, userid, out);

  basket_kernel<<<B_DIM, 256, 0, stream>>>(alpha, rho, lam, beta, theta, gamma,
                                           prices, price_table, item_ids, itemset,
                                           userid, out);
}
